// GroupedQueryAttention_79937931313668
// MI455X (gfx1250) — compile-verified, matrix operand order checked by reading, not run on hardware
//
#include <hip/hip_runtime.h>
#include <hip/hip_bf16.h>

typedef __attribute__((ext_vector_type(16))) __bf16 v16bf;
typedef __attribute__((ext_vector_type(8)))  __bf16 v8bf;
typedef __attribute__((ext_vector_type(8)))  float  v8f;

#define WMMA_BF16(a, b, c) \
  __builtin_amdgcn_wmma_f32_16x16x32_bf16(false, (a), false, (b), (short)0, (c), false, false)

constexpr int Bb  = 2;
constexpr int Ss  = 2048;
constexpr int Dd  = 2048;
constexpr int Hh  = 16;
constexpr int HKV = 2;
constexpr int DH  = 128;
constexpr int Gg  = Hh / HKV;       // 8
constexpr int MT  = Bb * Ss;        // 4096 tokens

// Load one 16x32 (bf16) WMMA fragment for this lane.
// ISA layout: lanes 0-15 hold K={0..7,16..23}, lanes 16-31 hold K={8..15,24..31}.
// `p` must already point at this lane's row + (half*8) element offset.
__device__ __forceinline__ v16bf load_frag(const __bf16* p) {
  union { v16bf v; v8bf h[2]; } u;
  u.h[0] = *(const v8bf*)(p);
  u.h[1] = *(const v8bf*)(p + 16);
  return u.v;
}

// CDNA5 async copy: global -> LDS, 16B per lane, tracked by ASYNCcnt.
// LDS byte address = low 32 bits of the generic pointer (flat aperture rule).
__device__ __forceinline__ void async_load_b128(const __bf16* gptr, __bf16* lptr) {
  unsigned la = (unsigned)(uintptr_t)lptr;
  asm volatile("global_load_async_to_lds_b128 %0, %1, off"
               :: "v"(la), "v"(gptr) : "memory");
}
__device__ __forceinline__ void wait_async0() {
  asm volatile("s_wait_asynccnt 0" ::: "memory");
}

// ---------------------------------------------------------------------------
// fp32 -> bf16 conversion
// ---------------------------------------------------------------------------
__global__ void cvt_f32_bf16(const float* __restrict__ in, __bf16* __restrict__ out, int n) {
  int i = blockIdx.x * blockDim.x + threadIdx.x;
  int stride = gridDim.x * blockDim.x;
  for (; i < n; i += stride) out[i] = (__bf16)in[i];
}

// ---------------------------------------------------------------------------
// C[M,N] = A[M,K] @ W[N,K]^T + bias   (torch Linear)
// bf16 inputs, f32 WMMA accumulation. 128x128 macro-tile, 8 waves (4x2),
// each wave computes a 32x64 region as 2x4 16x16 WMMA accumulators.
// Tile staging uses GLOBAL_LOAD_ASYNC_TO_LDS_B128 with LDS double buffering.
// OUT_TRANS stores C^T (used to produce V^T for the PV matmul).
// ---------------------------------------------------------------------------
template<bool OUT_F32, bool OUT_TRANS>
__global__ __launch_bounds__(256, 2)
void gemm_xwT(const __bf16* __restrict__ A, const __bf16* __restrict__ W,
              const float* __restrict__ bias, void* __restrict__ Cv,
              int M, int N, int K) {
  constexpr int LD = 40;  // padded row stride (elements), keeps 16B alignment
  __shared__ __bf16 sA[2][128 * LD];
  __shared__ __bf16 sB[2][128 * LD];

  const int tid  = threadIdx.x;
  const int lane = tid & 31;
  const int wid  = tid >> 5;
  const int wm   = wid & 3;         // 4 waves along M: 32 rows each
  const int wn   = wid >> 2;        // 2 waves along N: 64 cols each
  const int m0   = blockIdx.y * 128;
  const int n0   = blockIdx.x * 128;
  const int half = lane >> 4;
  const int r16  = lane & 15;
  const int fo   = half * 8;

  // Stage one 128x32 A tile + 128x32 W tile: 512 16B chunks each,
  // 2 chunks of each per thread, all as async LDS loads.
  auto stage = [&](int buf, int kt) {
    #pragma unroll
    for (int t = 0; t < 2; ++t) {
      const int c   = tid + t * 256;
      const int row = c >> 2;
      const int cc  = (c & 3) * 8;
      async_load_b128(A + (size_t)(m0 + row) * K + kt + cc, &sA[buf][row * LD + cc]);
      async_load_b128(W + (size_t)(n0 + row) * K + kt + cc, &sB[buf][row * LD + cc]);
    }
  };

  v8f acc[2][4] = {};

  stage(0, 0);
  wait_async0();
  __syncthreads();

  int buf = 0;
  for (int kt = 0; kt < K; kt += 32) {
    if (kt + 32 < K) stage(buf ^ 1, kt + 32);   // prefetch next K-tile async

    v16bf a0 = load_frag(&sA[buf][(wm * 32 +      r16) * LD + fo]);
    v16bf a1 = load_frag(&sA[buf][(wm * 32 + 16 + r16) * LD + fo]);
    #pragma unroll
    for (int ni = 0; ni < 4; ++ni) {
      v16bf bf = load_frag(&sB[buf][(wn * 64 + ni * 16 + r16) * LD + fo]);
      acc[0][ni] = WMMA_BF16(a0, bf, acc[0][ni]);
      acc[1][ni] = WMMA_BF16(a1, bf, acc[1][ni]);
    }

    wait_async0();       // this wave's async stores to LDS have landed
    __syncthreads();     // everyone's have landed; safe to read / reuse buffers
    buf ^= 1;
  }

  // C-layout: VGPR j, lane L -> row = j + 8*(L>=16), col = L&15
  #pragma unroll
  for (int ni = 0; ni < 4; ++ni) {
    const int ncol = n0 + wn * 64 + ni * 16 + r16;
    const float bv = bias[ncol];
    #pragma unroll
    for (int mi = 0; mi < 2; ++mi) {
      #pragma unroll
      for (int j = 0; j < 8; ++j) {
        const int mrow = m0 + wm * 32 + mi * 16 + j + half * 8;
        const float val = acc[mi][ni][j] + bv;
        if constexpr (OUT_TRANS) {
          ((__bf16*)Cv)[(size_t)ncol * M + mrow] = (__bf16)val;
        } else if constexpr (OUT_F32) {
          ((float*)Cv)[(size_t)mrow * N + ncol] = val;
        } else {
          ((__bf16*)Cv)[(size_t)mrow * N + ncol] = (__bf16)val;
        }
      }
    }
  }
}

// ---------------------------------------------------------------------------
// Flash-style causal GQA attention.
// One wave owns a 16-row query tile for one (b, h); loops over 32-key tiles.
// Q [B*S, H*DH] bf16, K [B*S, HKV*DH] bf16, Vt [HKV*DH, B*S] bf16 (transposed).
// Ctx [B*S, H*DH] bf16.
// ---------------------------------------------------------------------------
__global__ __launch_bounds__(128)
void gqa_attn(const __bf16* __restrict__ Q, const __bf16* __restrict__ Kp,
              const __bf16* __restrict__ Vt, __bf16* __restrict__ Ctx) {
  constexpr int LP = 40;                  // padded P-scratch row stride
  __shared__ __bf16 sP[4 * 16 * LP];      // per-wave 16x32 P transpose scratch

  const int tid  = threadIdx.x;
  const int lane = tid & 31;
  const int wid  = tid >> 5;
  const int half = lane >> 4;
  const int r16  = lane & 15;
  const int fo   = half * 8;

  const int qt = blockIdx.x * 4 + wid;    // 16-row query tile index
  const int h  = blockIdx.y, b = blockIdx.z;
  const int kv = h / Gg;
  const float inv_scale = 0.08838834764831845f;  // 1/sqrt(128)

  // Q fragments: 16 x 128, as 4 K-chunks of 32
  v16bf qa[4];
  {
    const __bf16* qb = Q + (size_t)(b * Ss + qt * 16 + r16) * (Hh * DH) + h * DH;
    #pragma unroll
    for (int c = 0; c < 4; ++c) qa[c] = load_frag(qb + c * 32 + fo);
  }

  float m_i[8], l_i[8];
  v8f o[8] = {};
  #pragma unroll
  for (int j = 0; j < 8; ++j) { m_i[j] = -__builtin_inff(); l_i[j] = 0.f; }

  __bf16* myP = &sP[wid * 16 * LP];
  const int jmax = (qt * 16 + 15) >> 5;   // last key tile touching the diagonal

  for (int jt = 0; jt <= jmax; ++jt) {
    const int kb = jt * 32;

    // S = Q @ K^T for 32 keys: two 16-key subtiles, K=DH chunked by 32
    v8f s0 = {}, s1 = {};
    #pragma unroll
    for (int c = 0; c < 4; ++c) {
      const __bf16* kp0 = Kp + (size_t)(b * Ss + kb + r16) * (HKV * DH) + kv * DH + c * 32 + fo;
      v16bf k0 = load_frag(kp0);
      v16bf k1 = load_frag(kp0 + (size_t)16 * (HKV * DH));
      s0 = WMMA_BF16(qa[c], k0, s0);
      s1 = WMMA_BF16(qa[c], k1, s1);
    }

    // scale + causal mask + online softmax (row = j + 8*half, col = lane&15)
    const int col0 = kb + r16, col1 = col0 + 16;
    #pragma unroll
    for (int j = 0; j < 8; ++j) {
      const int r = qt * 16 + j + half * 8;
      float x0 = (col0 <= r) ? s0[j] * inv_scale : -__builtin_inff();
      float x1 = (col1 <= r) ? s1[j] * inv_scale : -__builtin_inff();
      float mr = fmaxf(x0, x1);
      mr = fmaxf(mr, __shfl_xor(mr, 1));
      mr = fmaxf(mr, __shfl_xor(mr, 2));
      mr = fmaxf(mr, __shfl_xor(mr, 4));
      mr = fmaxf(mr, __shfl_xor(mr, 8));   // row-max within 16-lane group
      const float mnew = fmaxf(m_i[j], mr);
      const float corr = __expf(m_i[j] - mnew);
      const float p0 = __expf(x0 - mnew);
      const float p1 = __expf(x1 - mnew);
      float ps = p0 + p1;
      ps += __shfl_xor(ps, 1);
      ps += __shfl_xor(ps, 2);
      ps += __shfl_xor(ps, 4);
      ps += __shfl_xor(ps, 8);             // row-sum
      l_i[j] = l_i[j] * corr + ps;
      m_i[j] = mnew;
      #pragma unroll
      for (int dc = 0; dc < 8; ++dc) o[dc][j] *= corr;
      const int prow = j + half * 8;       // transpose P through LDS
      myP[prow * LP + r16]      = (__bf16)p0;
      myP[prow * LP + 16 + r16] = (__bf16)p1;
    }
    asm volatile("s_wait_dscnt 0" ::: "memory");  // wave-local DS store->load

    // O += P @ V  : P as A-fragment (K=32 keys), V^T rows give contiguous B-frags
    v16bf pa = load_frag(&myP[r16 * LP + fo]);
    #pragma unroll
    for (int dc = 0; dc < 8; ++dc) {
      const __bf16* vp = Vt + (size_t)(kv * DH + dc * 16 + r16) * MT + (size_t)b * Ss + kb + fo;
      v16bf vb = load_frag(vp);
      o[dc] = WMMA_BF16(pa, vb, o[dc]);
    }
  }

  // normalize and write context
  #pragma unroll
  for (int j = 0; j < 8; ++j) {
    const float invl = 1.0f / l_i[j];
    const size_t row = (size_t)b * Ss + qt * 16 + j + half * 8;
    #pragma unroll
    for (int dc = 0; dc < 8; ++dc)
      Ctx[row * (Hh * DH) + h * DH + dc * 16 + r16] = (__bf16)(o[dc][j] * invl);
  }
}

// ---------------------------------------------------------------------------
extern "C" void kernel_launch(void* const* d_in, const int* in_sizes, int n_in,
                              void* d_out, int out_size, void* d_ws, size_t ws_size,
                              hipStream_t stream) {
  (void)in_sizes; (void)n_in; (void)out_size; (void)ws_size;
  const float* x  = (const float*)d_in[0];
  const float* Wq = (const float*)d_in[1];
  const float* bq = (const float*)d_in[2];
  const float* Wk = (const float*)d_in[3];
  const float* bk = (const float*)d_in[4];
  const float* Wv = (const float*)d_in[5];
  const float* bv = (const float*)d_in[6];
  const float* Wo = (const float*)d_in[7];
  const float* bo = (const float*)d_in[8];
  float* out = (float*)d_out;

  char* ws = (char*)d_ws;
  size_t off = 0;
  auto carve = [&](size_t elems) {
    __bf16* p = (__bf16*)(ws + off);
    off += ((elems * sizeof(__bf16)) + 255) & ~(size_t)255;
    return p;
  };
  __bf16* xb  = carve((size_t)MT * Dd);        // x in bf16
  __bf16* wqb = carve((size_t)Dd * Dd);
  __bf16* wkb = carve((size_t)HKV * DH * Dd);
  __bf16* wvb = carve((size_t)HKV * DH * Dd);
  __bf16* wob = carve((size_t)Dd * Dd);
  __bf16* Qb  = carve((size_t)MT * Dd);        // Q [B*S, H*DH]
  __bf16* Kb  = carve((size_t)MT * HKV * DH);  // K [B*S, HKV*DH]
  __bf16* Vtb = carve((size_t)MT * HKV * DH);  // V^T [HKV*DH, B*S]
  __bf16* Cxb = carve((size_t)MT * Dd);        // attention context

  const dim3 cb(256), cg(1024);
  cvt_f32_bf16<<<cg, cb, 0, stream>>>(x,  xb,  MT * Dd);
  cvt_f32_bf16<<<cg, cb, 0, stream>>>(Wq, wqb, Dd * Dd);
  cvt_f32_bf16<<<cg, cb, 0, stream>>>(Wk, wkb, HKV * DH * Dd);
  cvt_f32_bf16<<<cg, cb, 0, stream>>>(Wv, wvb, HKV * DH * Dd);
  cvt_f32_bf16<<<cg, cb, 0, stream>>>(Wo, wob, Dd * Dd);

  gemm_xwT<false, false><<<dim3(Dd / 128,       MT / 128), 256, 0, stream>>>(xb, wqb, bq, Qb,  MT, Dd,       Dd);
  gemm_xwT<false, false><<<dim3((HKV*DH) / 128, MT / 128), 256, 0, stream>>>(xb, wkb, bk, Kb,  MT, HKV * DH, Dd);
  gemm_xwT<false, true ><<<dim3((HKV*DH) / 128, MT / 128), 256, 0, stream>>>(xb, wvb, bv, Vtb, MT, HKV * DH, Dd);

  gqa_attn<<<dim3(Ss / 16 / 4, Hh, Bb), 128, 0, stream>>>(Qb, Kb, Vtb, Cxb);

  gemm_xwT<true, false><<<dim3(Dd / 128, MT / 128), 256, 0, stream>>>(Cxb, wob, bo, out, MT, Dd, Dd);
}
